// NMRNN_25039659336107
// MI455X (gfx1250) — matrix operand order repaired and checked by reading, not compiled, not run on hardware
//
#include <hip/hip_runtime.h>
#include <hip/hip_bf16.h>
#include <cstdint>
#include <cstddef>

// ---------------- CDNA5 WMMA types ----------------
typedef __bf16  bf16_t;
typedef bf16_t  v16bf __attribute__((ext_vector_type(16)));
typedef bf16_t  v8bf  __attribute__((ext_vector_type(8)));
typedef float   v8f   __attribute__((ext_vector_type(8)));
typedef float   v4f   __attribute__((ext_vector_type(4)));

#define T_STEPS 1024
#define BATCH   512
#define DIMZ    256
#define DIMH    256

// ---------------- LDS layout (bytes) ----------------
// Weights, bf16, stored [N][K] row-major = natural [out,in] layout so that a
// B-fragment is two contiguous 16B LDS loads per lane (ISA 7.12.2 B layout:
// lanes 0-15 hold K=kb..kb+7 & kb+16..kb+23, lanes 16-31 hold K=kb+8..kb+15 & kb+24..kb+31, lane&15 = N).
#define WZZ_OFF    0          // [256][256]
#define WZIN_OFF   131072     // [256][64]
#define WIN_OFF    163840     // [256][32]
#define WZOUT_OFF  180224     // [16][256]  rows 8..15 zero
#define WOUT_OFF   188416     // [16][256]
#define U_OFF      196608     // [256][32]  cols 8..31 zero
#define VT_OFF     212992     // [16][256]  rows 8..15 zero (V transposed)
#define BZIN_OFF   221184     // f32 [256]
#define BIN_OFF    222208     // f32 [256]
#define BZOUT_OFF  223232     // f32 [16] (pad)
#define BOUT_OFF   223296     // f32 [16]
#define PW_OFF     223360     // per-wave activation buffers
#define PW_STRIDE  25600      // tanh_z 8K | tanh_h 8K | z_bf16 8K | gs 1K
#define LDS_BYTES  (PW_OFF + 4*PW_STRIDE)   // 325760 <= 327680 (320KB/WGP)

__device__ __forceinline__ v8f wmma_bf16(v16bf a, v16bf b, v8f c) {
  return __builtin_amdgcn_wmma_f32_16x16x32_bf16(false, a, false, b, (short)0, c,
                                                 false, false);
}

union AFragU { v16bf v; v8bf h[2]; };

// A-fragment (16x32 bf16) from row-major LDS [16][ldk].
// ISA 7.12.2: lane&15 = M row; lanes<16 hold K=kb..kb+7 & kb+16..kb+23,
// lanes>=16 hold K=kb+8..kb+15 & kb+24..kb+31.  Two 16B loads per lane.
__device__ __forceinline__ v16bf ldsA(const bf16_t* base, int ldk, int kb, int lane) {
  const int row = lane & 15;
  const int ko  = (lane >> 1) & 8;          // 0 or 8
  const bf16_t* p = base + row * ldk + kb + ko;
  AFragU f;
  f.h[0] = *(const v8bf*)(p);
  f.h[1] = *(const v8bf*)(p + 16);
  return f.v;
}

// B-fragment (32x16 bf16) from LDS weight stored [N][K] row-major.
__device__ __forceinline__ v16bf ldsB(const bf16_t* base, int ldk, int nb, int kb,
                                      int lane) {
  const int n  = lane & 15;
  const int ko = (lane >> 1) & 8;           // 0 or 8
  const bf16_t* p = base + (nb + n) * ldk + kb + ko;
  AFragU f;
  f.h[0] = *(const v8bf*)(p);
  f.h[1] = *(const v8bf*)(p + 16);
  return f.v;
}

// A-fragment built from a row-major f32 global matrix [16][32] (s_t or c_t tile).
__device__ __forceinline__ v16bf gA32(const float* base, int lane) {
  const int row = lane & 15;
  const int ko  = (lane >> 1) & 8;          // 0 or 8
  const float* p = base + row * 32 + ko;
  v4f a0 = *(const v4f*)(p);
  v4f a1 = *(const v4f*)(p + 4);
  v4f b0 = *(const v4f*)(p + 16);
  v4f b1 = *(const v4f*)(p + 20);
  v16bf r;
  #pragma unroll
  for (int i = 0; i < 4; ++i) {
    r[i]      = (bf16_t)a0[i];
    r[i + 4]  = (bf16_t)a1[i];
    r[i + 8]  = (bf16_t)b0[i];
    r[i + 12] = (bf16_t)b1[i];
  }
  return r;
}

// Fast transcendentals: one v_exp_f32 + one v_rcp_f32 each (no IEEE div chain).
// tanh: rcp(inf)=0 -> +1, rcp(1)=1 -> -1, so saturation at +/-inf is exact.
__device__ __forceinline__ float fast_tanh(float x) {
  return 1.0f - 2.0f * __builtin_amdgcn_rcpf(__expf(2.0f * x) + 1.0f);
}
__device__ __forceinline__ float fast_sigmoid(float x) {
  return __builtin_amdgcn_rcpf(1.0f + __expf(-x));
}

__global__ __launch_bounds__(128, 1)
void nmrnn_persistent(const float* __restrict__ s, const float* __restrict__ c,
                      const float* __restrict__ Wzz, const float* __restrict__ Wzin,
                      const float* __restrict__ bzin, const float* __restrict__ Wzout,
                      const float* __restrict__ bzout, const float* __restrict__ U,
                      const float* __restrict__ V, const float* __restrict__ Win,
                      const float* __restrict__ bin, const float* __restrict__ Wout,
                      const float* __restrict__ bout,
                      float* __restrict__ out, float* __restrict__ states) {
  extern __shared__ char smem[];
  bf16_t* wzz   = (bf16_t*)(smem + WZZ_OFF);
  bf16_t* wzin  = (bf16_t*)(smem + WZIN_OFF);
  bf16_t* win   = (bf16_t*)(smem + WIN_OFF);
  bf16_t* wzout = (bf16_t*)(smem + WZOUT_OFF);
  bf16_t* wout  = (bf16_t*)(smem + WOUT_OFF);
  bf16_t* ulds  = (bf16_t*)(smem + U_OFF);
  bf16_t* vt    = (bf16_t*)(smem + VT_OFF);
  float*  bzinL  = (float*)(smem + BZIN_OFF);
  float*  binL   = (float*)(smem + BIN_OFF);
  float*  bzoutL = (float*)(smem + BZOUT_OFF);
  float*  boutL  = (float*)(smem + BOUT_OFF);

  const int tid = threadIdx.x;

  // ---- one-time weight convert f32 -> bf16 into WMMA-friendly LDS layouts ----
  for (int i = tid; i < 256 * 256; i += 128) wzz[i] = (bf16_t)Wzz[i];
  for (int i = tid; i < 256 * 64;  i += 128) wzin[i] = (bf16_t)Wzin[i];
  for (int i = tid; i < 256 * 32;  i += 128) win[i] = (bf16_t)Win[i];
  for (int i = tid; i < 16 * 256;  i += 128) {
    int n = i >> 8, k = i & 255;
    wzout[i] = (n < 8) ? (bf16_t)Wzout[n * 256 + k] : (bf16_t)0.0f;
    wout[i]  = (bf16_t)Wout[i];
    vt[i]    = (n < 8) ? (bf16_t)V[k * 8 + n] : (bf16_t)0.0f;   // V transposed
  }
  for (int i = tid; i < 256 * 32; i += 128) {
    int n = i >> 5, k = i & 31;
    ulds[i] = (k < 8) ? (bf16_t)U[n * 8 + k] : (bf16_t)0.0f;    // K padded to 32
  }
  for (int i = tid; i < 256; i += 128) { bzinL[i] = bzin[i]; binL[i] = bin[i]; }
  if (tid < 16) { bzoutL[tid] = (tid < 8) ? bzout[tid] : 0.0f; boutL[tid] = bout[tid]; }
  // zero per-wave activation buffers (t=0 state is zero; gs K-pad stays zero)
  for (int i = tid; i < (4 * PW_STRIDE) / 4; i += 128)
    ((uint32_t*)(smem + PW_OFF))[i] = 0u;
  __syncthreads();   // only inter-wave sync needed: weights are read-only after this

  const int wave = tid >> 5;
  const int lane = tid & 31;
  const int b0   = blockIdx.x * 64 + wave * 16;   // 16 batch rows per wave
  const int col  = lane & 15;                      // D-layout column
  const int rb   = (lane >> 1) & 8;                // D-layout row base (0 or 8)

  bf16_t* tz  = (bf16_t*)(smem + PW_OFF + wave * PW_STRIDE); // tanh(z) [16][256]
  bf16_t* th  = tz + 4096;                                   // tanh(h) [16][256]
  bf16_t* zb  = th + 4096;                                   // z bf16  [16][256]
  bf16_t* gsb = zb + 4096;                                   // g*sig   [16][32]

  for (int t = 0; t < T_STEPS; ++t) {
    const float* sP = s + ((size_t)t * BATCH + b0) * 32;
    const float* cP = c + ((size_t)t * BATCH + b0) * 32;
    float* stO = states + ((size_t)t * BATCH + b0) * 512;
    const float* stI = states + ((size_t)(t - 1) * BATCH + b0) * 512;
    float* oP  = out + ((size_t)t * BATCH + b0) * 16;

    if (t + 1 < T_STEPS) {  // pull next step's inputs toward L2 (global_prefetch_b8)
      __builtin_prefetch(sP + (size_t)BATCH * 32 + col * 32, 0, 1);
      __builtin_prefetch(cP + (size_t)BATCH * 32 + col * 32, 0, 1);
    }

    // input fragments: x = [s_t | c_t], K = 0..31 -> s, 32..63 -> c
    v16bf axs = gA32(sP, lane);
    v16bf axc = gA32(cP, lane);
    // A fragments of tanh(z_{t-1}); full unroll below keeps these VGPR-resident
    v16bf az[8];
    #pragma unroll
    for (int k = 0; k < 8; ++k) az[k] = ldsA(tz, 256, k * 32, lane);

    // ---------------- z update: tmp = tanh(z)@Wzz.T + x@Wzin.T + bz ----------------
    #pragma unroll
    for (int n = 0; n < 16; ++n) {
      v8f acc = {};
      #pragma unroll
      for (int k = 0; k < 8; ++k)
        acc = wmma_bf16(az[k], ldsB(wzz, 256, n * 16, k * 32, lane), acc);
      acc = wmma_bf16(axs, ldsB(wzin, 64, n * 16, 0, lane), acc);
      acc = wmma_bf16(axc, ldsB(wzin, 64, n * 16, 32, lane), acc);
      const float bias = bzinL[n * 16 + col];
      #pragma unroll
      for (int r = 0; r < 8; ++r) {
        const int ro = (rb + r);
        const int co = n * 16 + col;
        const float zo = (t > 0) ? stI[ro * 512 + co] : 0.0f;   // L2-resident
        const float zn = 0.99f * zo + 0.01f * (acc[r] + bias);
        stO[ro * 512 + co] = zn;                     // states output (z half)
        zb[ro * 256 + co]  = (bf16_t)zn;             // A-operand for signal GEMM
        tz[ro * 256 + co]  = (bf16_t)fast_tanh(zn);  // A-operand for next step
      }
    }

    // ---------------- signal = sigmoid(z@Wzout.T + b);  g = tanh(h)@V --------------
    v8f sigacc = {}, gacc = {};
    #pragma unroll
    for (int k = 0; k < 8; ++k) {
      sigacc = wmma_bf16(ldsA(zb, 256, k * 32, lane),
                         ldsB(wzout, 256, 0, k * 32, lane), sigacc);
      gacc   = wmma_bf16(ldsA(th, 256, k * 32, lane),
                         ldsB(vt, 256, 0, k * 32, lane), gacc);
    }
    {
      const float bzo = bzoutL[col];
      #pragma unroll
      for (int r = 0; r < 8; ++r) {
        const float sg = fast_sigmoid(sigacc[r] + bzo);
        const float gv = gacc[r] * sg;
        if (col < 8) gsb[(rb + r) * 32 + col] = (bf16_t)gv;  // cols 8..31 stay 0
      }
    }

    // ---------------- h update: tmp = gs@U.T + s@Win.T + b -------------------------
    v16bf ag = ldsA(gsb, 32, 0, lane);   // [16,32] with K>=8 zero
    #pragma unroll
    for (int n = 0; n < 16; ++n) {
      v8f acc = {};
      acc = wmma_bf16(ag,  ldsB(ulds, 32, n * 16, 0, lane), acc);
      acc = wmma_bf16(axs, ldsB(win,  32, n * 16, 0, lane), acc);
      const float bias = binL[n * 16 + col];
      #pragma unroll
      for (int r = 0; r < 8; ++r) {
        const int ro = (rb + r);
        const int co = n * 16 + col;
        const float ho = (t > 0) ? stI[ro * 512 + 256 + co] : 0.0f;
        const float hn = 0.9f * ho + 0.1f * (acc[r] + bias);
        stO[ro * 512 + 256 + co] = hn;               // states output (h half)
        th[ro * 256 + co] = (bf16_t)fast_tanh(hn);   // feeds out-GEMM & next step
      }
    }

    // ---------------- out = tanh(h_new)@Wout.T + b_out -----------------------------
    v8f oacc = {};
    #pragma unroll
    for (int k = 0; k < 8; ++k)
      oacc = wmma_bf16(ldsA(th, 256, k * 32, lane),
                       ldsB(wout, 256, 0, k * 32, lane), oacc);
    {
      const float bo = boutL[col];
      #pragma unroll
      for (int r = 0; r < 8; ++r) oP[(rb + r) * 16 + col] = oacc[r] + bo;
    }
  }
}

extern "C" void kernel_launch(void* const* d_in, const int* in_sizes, int n_in,
                              void* d_out, int out_size, void* d_ws, size_t ws_size,
                              hipStream_t stream) {
  const float* s     = (const float*)d_in[0];
  const float* c     = (const float*)d_in[1];
  const float* Wzz   = (const float*)d_in[2];
  const float* Wzin  = (const float*)d_in[3];
  const float* bzin  = (const float*)d_in[4];
  const float* Wzout = (const float*)d_in[5];
  const float* bzout = (const float*)d_in[6];
  const float* U     = (const float*)d_in[7];
  const float* V     = (const float*)d_in[8];
  const float* Win   = (const float*)d_in[9];
  const float* bin   = (const float*)d_in[10];
  const float* Wout  = (const float*)d_in[11];
  const float* bout  = (const float*)d_in[12];

  float* outp   = (float*)d_out;                               // [T,B,16]
  float* states = outp + (size_t)T_STEPS * BATCH * 16;         // [T,B,512]

  hipFuncSetAttribute((const void*)nmrnn_persistent,
                      hipFuncAttributeMaxDynamicSharedMemorySize, LDS_BYTES);

  // 8 workgroups x 4 waves x 16 batch rows = 512 = B. Whole T-loop in-kernel.
  nmrnn_persistent<<<8, 128, LDS_BYTES, stream>>>(
      s, c, Wzz, Wzin, bzin, Wzout, bzout, U, V, Win, bin, Wout, bout,
      outp, states);
}
